// I2MoE_28157805592787
// MI455X (gfx1250) — matrix operand order repaired and checked by reading, not compiled, not run on hardware
//
#include <hip/hip_runtime.h>
#include <hip/hip_bf16.h>
#include <math.h>

// ---------------------------------------------------------------------------
// Types for gfx1250 WMMA (wave32)
// ---------------------------------------------------------------------------
typedef __bf16 bf16_t;
typedef __attribute__((ext_vector_type(16))) __bf16 v16bf;
typedef __attribute__((ext_vector_type(8)))  float  v8f;

#define BM 64
#define BN 64
#define BK 32

enum { EPI_NONE = 0, EPI_SILU = 1, EPI_GELU = 2, EPI_SOFTPLUS = 3, EPI_RELU = 4 };

template <int EPI>
__device__ __forceinline__ float epilogue(float x) {
    if constexpr (EPI == EPI_SILU)      return x / (1.f + __expf(-x));
    else if constexpr (EPI == EPI_GELU) return 0.5f * x * (1.f + erff(x * 0.7071067811865476f));
    else if constexpr (EPI == EPI_SOFTPLUS) return (x > 20.f) ? x : log1pf(__expf(x));
    else if constexpr (EPI == EPI_RELU) return x > 0.f ? x : 0.f;
    else return x;
}

// A-fragment, 16x32 bf16 tile in LDS, row stride BK.
// ISA 7.12.2: lane m=lane&15; lanes<16: elems0-7 => K0..7, elems8-15 => K16..23
//             lanes>=16: elems0-7 => K8..15, elems8-15 => K24..31
__device__ __forceinline__ v16bf frag_a(const bf16_t* tile, int lane) {
    const bf16_t* p = tile + (lane & 15) * BK;
    const int b0 = (lane & 16) ? 8 : 0;
    const int b1 = (lane & 16) ? 24 : 16;
    v16bf f;
#pragma unroll
    for (int j = 0; j < 8; ++j) { f[j] = p[b0 + j]; f[j + 8] = p[b1 + j]; }
    return f;
}

// B-fragment: LDS holds W tile as [n][k] (row stride BK). lane n=lane&15;
// lanes<16: elems0-7 => K0..7, elems8-15 => K16..23; lanes>=16: K8..15, K24..31 (same as A).
__device__ __forceinline__ v16bf frag_b(const bf16_t* tile, int lane) {
    const bf16_t* p = tile + (lane & 15) * BK;
    const int b0 = (lane & 16) ? 8 : 0;
    const int b1 = (lane & 16) ? 24 : 16;
    v16bf f;
#pragma unroll
    for (int j = 0; j < 8; ++j) { f[j] = p[b0 + j]; f[j + 8] = p[b1 + j]; }
    return f;
}

// ---------------------------------------------------------------------------
// Generic WMMA GEMM: C[M,N] = epi(A[M,K] * W[N,K]^T + bias[N])
// A fp32 (lda), W fp32 row-major [N,K], C fp32 (ldc). bf16 operands, f32 acc.
// REQUIRES K % 32 == 0 (true for every projection in this model).
// Block: 128 threads = 4 waves; block tile 64x64; wave tile 32x32 (2x2 WMMA).
// Branchless staging (clamped rows + zero-select) and LDS double-buffering:
// one barrier per K-step, no divergent exec-mask paths in the hot loop.
// ---------------------------------------------------------------------------
template <int EPI>
__global__ __launch_bounds__(128) void gemm_wmma(
    const float* __restrict__ A, int lda,
    const float* __restrict__ W,
    const float* __restrict__ bias,
    float* __restrict__ C, int ldc,
    int M, int N, int K)
{
    __shared__ __align__(16) bf16_t sA[2][BM][BK];
    __shared__ __align__(16) bf16_t sB[2][BN][BK];

    const int tid  = threadIdx.x;
    const int lane = tid & 31;
    const int wave = tid >> 5;
    const int m0 = blockIdx.y * BM;
    const int n0 = blockIdx.x * BN;
    const int wm = (wave >> 1) * 32;
    const int wn = (wave & 1) * 32;

    v8f acc[2][2] = {};

    // staging assignment: thread stages row r, k-halves [kb, kb+16)
    const int r  = tid >> 1;          // 0..63
    const int kb = (tid & 1) * 16;    // 0 or 16
    const int gm = m0 + r;
    const int gn = n0 + r;
    const bool aOk = gm < M;
    const bool bOk = gn < N;
    // clamp to a valid row so loads are always in-bounds; zero-select after.
    const float* Arow = A + (size_t)(aOk ? gm : (M - 1)) * lda + kb;
    const float* Wrow = W + (size_t)(bOk ? gn : (N - 1)) * K + kb;
    const float am = aOk ? 1.f : 0.f;
    const float bm = bOk ? 1.f : 0.f;

    auto stage = [&](int kk, int buf) {
        const float* pa = Arow + kk;
        const float* pw = Wrow + kk;
        bf16_t* da = &sA[buf][r][kb];
        bf16_t* db = &sB[buf][r][kb];
#pragma unroll
        for (int v = 0; v < 4; ++v) {
            float4 qa = *reinterpret_cast<const float4*>(pa + v * 4);
            float4 qb = *reinterpret_cast<const float4*>(pw + v * 4);
            da[v * 4 + 0] = (bf16_t)(qa.x * am);
            da[v * 4 + 1] = (bf16_t)(qa.y * am);
            da[v * 4 + 2] = (bf16_t)(qa.z * am);
            da[v * 4 + 3] = (bf16_t)(qa.w * am);
            db[v * 4 + 0] = (bf16_t)(qb.x * bm);
            db[v * 4 + 1] = (bf16_t)(qb.y * bm);
            db[v * 4 + 2] = (bf16_t)(qb.z * bm);
            db[v * 4 + 3] = (bf16_t)(qb.w * bm);
        }
    };

    stage(0, 0);
    __syncthreads();

    int buf = 0;
    for (int k0 = 0; k0 < K; k0 += BK, buf ^= 1) {
        if (k0 + BK < K) {
            __builtin_prefetch(Arow + k0 + 2 * BK, 0, 1);
            __builtin_prefetch(Wrow + k0 + 2 * BK, 0, 1);
            stage(k0 + BK, buf ^ 1);
        }

        v16bf a0 = frag_a(&sA[buf][wm][0], lane);
        v16bf a1 = frag_a(&sA[buf][wm + 16][0], lane);
        v16bf b0 = frag_b(&sB[buf][wn][0], lane);
        v16bf b1 = frag_b(&sB[buf][wn + 16][0], lane);

        acc[0][0] = __builtin_amdgcn_wmma_f32_16x16x32_bf16(false, a0, false, b0, (short)0, acc[0][0], false, false);
        acc[0][1] = __builtin_amdgcn_wmma_f32_16x16x32_bf16(false, a0, false, b1, (short)0, acc[0][1], false, false);
        acc[1][0] = __builtin_amdgcn_wmma_f32_16x16x32_bf16(false, a1, false, b0, (short)0, acc[1][0], false, false);
        acc[1][1] = __builtin_amdgcn_wmma_f32_16x16x32_bf16(false, a1, false, b1, (short)0, acc[1][1], false, false);
        __syncthreads();
    }

    // C/D layout: elem rr -> M = rr + 8*(lane>=16); N = lane&15
    const int cn    = lane & 15;
    const int mhalf = (lane & 16) ? 8 : 0;
#pragma unroll
    for (int i = 0; i < 2; ++i)
#pragma unroll
        for (int j = 0; j < 2; ++j) {
            int gcol = n0 + wn + j * 16 + cn;
            if (gcol >= N) continue;
            float bval = bias ? bias[gcol] : 0.f;
#pragma unroll
            for (int rr = 0; rr < 8; ++rr) {
                int grow = m0 + wm + i * 16 + mhalf + rr;
                if (grow < M)
                    C[(size_t)grow * ldc + gcol] = epilogue<EPI>(acc[i][j][rr] + bval);
            }
        }
}

// ---------------------------------------------------------------------------
// Element-wise / small kernels
// ---------------------------------------------------------------------------
__global__ __launch_bounds__(256) void gather_slice(
    float* __restrict__ dst, const float* __restrict__ x, int s0, int L, int D)
{
    size_t idx = (size_t)blockIdx.x * blockDim.x + threadIdx.x;
    size_t total = (size_t)128 * L * D;
    if (idx >= total) return;
    int d = (int)(idx % D);
    size_t t = idx / D;
    int b = (int)(t / L), l = (int)(t % L);
    dst[idx] = x[((size_t)b * 39 + s0 + l) * D + d];
}

__global__ __launch_bounds__(256) void scatter_slice(
    float* __restrict__ base, const float* __restrict__ src, int s0, int L, int D)
{
    size_t idx = (size_t)blockIdx.x * blockDim.x + threadIdx.x;
    size_t total = (size_t)128 * L * D;
    if (idx >= total) return;
    int d = (int)(idx % D);
    size_t t = idx / D;
    int b = (int)(t / L), l = (int)(t % L);
    base[((size_t)b * 39 + s0 + l) * D + d] = src[idx];
}

__global__ __launch_bounds__(256) void flip_seq(
    float* __restrict__ dst, const float* __restrict__ src, int Bc, int L, int D)
{
    size_t idx = (size_t)blockIdx.x * blockDim.x + threadIdx.x;
    size_t total = (size_t)Bc * L * D;
    if (idx >= total) return;
    int d = (int)(idx % D);
    size_t t = idx / D;
    int b = (int)(t / L), l = (int)(t % L);
    dst[((size_t)b * L + (L - 1 - l)) * D + d] = src[idx];
}

// depthwise causal conv + bias + SiLU:  xc = silu(conv1d(xs) + cb)
__global__ __launch_bounds__(256) void dwconv_silu(
    const float* __restrict__ xs, const float* __restrict__ cw,
    const float* __restrict__ cb, float* __restrict__ xc,
    int Bc, int L, int d_in, int d_conv)
{
    size_t idx = (size_t)blockIdx.x * blockDim.x + threadIdx.x;
    size_t total = (size_t)Bc * L * d_in;
    if (idx >= total) return;
    int c = (int)(idx % d_in);
    size_t t = idx / d_in;
    int b = (int)(t / L), l = (int)(t % L);
    float acc = cb[c];
    for (int k = 0; k < d_conv; ++k) {
        int ls = l - (d_conv - 1) + k;
        if (ls >= 0)
            acc += xs[((size_t)b * L + ls) * d_in + c] * cw[(size_t)c * d_conv + k];
    }
    xc[idx] = acc / (1.f + __expf(-acc));
}

// selective scan: per (batch, channel) recurrence over L with 16 states
__global__ __launch_bounds__(256) void mamba_scan(
    const float* __restrict__ dtv, const float* __restrict__ dbl,
    const float* __restrict__ xc,  const float* __restrict__ z,
    const float* __restrict__ A_log, const float* __restrict__ Dp,
    float* __restrict__ y, int Bc, int L, int d_in)
{
    int idx = blockIdx.x * blockDim.x + threadIdx.x;
    if (idx >= Bc * d_in) return;
    int b = idx / d_in, c = idx % d_in;

    float Aarr[16];
#pragma unroll
    for (int s = 0; s < 16; ++s) Aarr[s] = -__expf(A_log[(size_t)c * 16 + s]);
    const float Df = Dp[c];

    float h[16];
#pragma unroll
    for (int s = 0; s < 16; ++s) h[s] = 0.f;

    for (int l = 0; l < L; ++l) {
        size_t t = (size_t)b * L + l;
        float dt = dtv[t * d_in + c];
        float xv = xc[t * d_in + c];
        const float* Bm = dbl + t * 96 + 64;
        const float* Cm = dbl + t * 96 + 80;
        float acc = 0.f;
#pragma unroll
        for (int s = 0; s < 16; ++s) {
            float dA = __expf(dt * Aarr[s]);
            h[s] = dA * h[s] + dt * xv * Bm[s];
            acc += h[s] * Cm[s];
        }
        float yy = acc + xv * Df;
        float zz = z[t * d_in + c];
        y[t * d_in + c] = yy * (zz / (1.f + __expf(-zz)));
    }
}

// out = [gelu](LN(a + b? + c?)) — one block per row, D <= 1024
template <int GELU>
__global__ __launch_bounds__(256) void add_ln(
    const float* __restrict__ a, const float* __restrict__ b,
    const float* __restrict__ c, const float* __restrict__ w,
    const float* __restrict__ bias, float* __restrict__ out, int D)
{
    __shared__ float buf[1024];
    __shared__ float red[256];
    const int row = blockIdx.x, tid = threadIdx.x;
    float lsum = 0.f;
    for (int d = tid; d < D; d += 256) {
        float v = a[(size_t)row * D + d];
        if (b) v += b[(size_t)row * D + d];
        if (c) v += c[(size_t)row * D + d];
        buf[d] = v;
        lsum += v;
    }
    red[tid] = lsum;
    __syncthreads();
    for (int s = 128; s > 0; s >>= 1) { if (tid < s) red[tid] += red[tid + s]; __syncthreads(); }
    const float mu = red[0] / (float)D;
    __syncthreads();
    float lvar = 0.f;
    for (int d = tid; d < D; d += 256) { float t = buf[d] - mu; lvar += t * t; }
    red[tid] = lvar;
    __syncthreads();
    for (int s = 128; s > 0; s >>= 1) { if (tid < s) red[tid] += red[tid + s]; __syncthreads(); }
    const float rstd = rsqrtf(red[0] / (float)D + 1e-5f);
    for (int d = tid; d < D; d += 256) {
        float v = (buf[d] - mu) * rstd * w[d] + bias[d];
        if constexpr (GELU) v = 0.5f * v * (1.f + erff(v * 0.7071067811865476f));
        out[(size_t)row * D + d] = v;
    }
}

// feats[b,e,:] = mean over L of base[b, s0:s0+L, :]
__global__ __launch_bounds__(256) void mean_slice(
    const float* __restrict__ base, float* __restrict__ feats,
    int e, int s0, int L, int D)
{
    int idx = blockIdx.x * blockDim.x + threadIdx.x;
    if (idx >= 128 * D) return;
    int b = idx / D, d = idx % D;
    float s = 0.f;
    for (int l = 0; l < L; ++l) s += base[((size_t)b * 39 + s0 + l) * D + d];
    feats[((size_t)b * 4 + e) * D + d] = s / (float)L;
}

// redundancy expert input: softmax(feats·featsᵀ) mixing, mean, broadcast over 39
__global__ __launch_bounds__(256) void red_prep(
    const float* __restrict__ feats, float* __restrict__ red_in, int D)
{
    const int b = blockIdx.x, tid = threadIdx.x;
    __shared__ float rw[4][4];
    if (tid < 16) {
        int i = tid / 4, j = tid % 4;
        const float* fi = feats + ((size_t)b * 4 + i) * D;
        const float* fj = feats + ((size_t)b * 4 + j) * D;
        float s = 0.f;
        for (int d = 0; d < D; ++d) s += fi[d] * fj[d];
        rw[i][j] = s;
    }
    __syncthreads();
    if (tid < 4) {
        float mx = rw[tid][0];
        for (int j = 1; j < 4; ++j) mx = fmaxf(mx, rw[tid][j]);
        float e[4], s = 0.f;
        for (int j = 0; j < 4; ++j) { e[j] = __expf(rw[tid][j] - mx); s += e[j]; }
        for (int j = 0; j < 4; ++j) rw[tid][j] = e[j] / s;
    }
    __syncthreads();
    for (int d = tid; d < D; d += 256) {
        float acc = 0.f;
        for (int i = 0; i < 4; ++i) {
            float r = 0.f;
            for (int j = 0; j < 4; ++j) r += rw[i][j] * feats[((size_t)b * 4 + j) * D + d];
            acc += r;
        }
        acc *= 0.25f;
        for (int l = 0; l < 39; ++l) red_in[((size_t)b * 39 + l) * D + d] = acc;
    }
}

// tiny MHA core: T=4, hd=128, one block per (b,h)
__global__ __launch_bounds__(128) void mha_attn(
    const float* __restrict__ qkv, float* __restrict__ outb, int D, int H, int hd)
{
    const int bh = blockIdx.x, b = bh / H, h = bh % H;
    const int tid = threadIdx.x;
    __shared__ float att[4][4];
    const float scale = rsqrtf((float)hd);
    const size_t rb = (size_t)b * 4;
    if (tid < 16) {
        int tq = tid / 4, tk = tid % 4;
        const float* q = qkv + (rb + tq) * 3 * D + h * hd;
        const float* k = qkv + (rb + tk) * 3 * D + D + h * hd;
        float s = 0.f;
        for (int d = 0; d < hd; ++d) s += q[d] * k[d];
        att[tq][tk] = s * scale;
    }
    __syncthreads();
    if (tid < 4) {
        float mx = att[tid][0];
        for (int j = 1; j < 4; ++j) mx = fmaxf(mx, att[tid][j]);
        float e[4], s = 0.f;
        for (int j = 0; j < 4; ++j) { e[j] = __expf(att[tid][j] - mx); s += e[j]; }
        for (int j = 0; j < 4; ++j) att[tid][j] = e[j] / s;
    }
    __syncthreads();
    if (tid < hd) {
        for (int tq = 0; tq < 4; ++tq) {
            float o = 0.f;
            for (int tk = 0; tk < 4; ++tk)
                o += att[tq][tk] * qkv[(rb + tk) * 3 * D + 2 * D + h * hd + tid];
            outb[(rb + tq) * D + h * hd + tid] = o;
        }
    }
}

// small-N dense: C[M,N] = A[M,K] * W[N,K]^T + bias
__global__ __launch_bounds__(256) void small_linear(
    const float* __restrict__ A, int lda, const float* __restrict__ W,
    const float* __restrict__ bias, float* __restrict__ C, int M, int N, int K)
{
    int i = blockIdx.x * blockDim.x + threadIdx.x;
    if (i >= M * N) return;
    int m = i / N, n = i % N;
    const float* a = A + (size_t)m * lda;
    const float* w = W + (size_t)n * K;
    float s = bias ? bias[n] : 0.f;
    for (int k = 0; k < K; ++k) s += a[k] * w[k];
    C[i] = s;
}

__global__ __launch_bounds__(128) void row_softmax(
    const float* __restrict__ in, float* __restrict__ out, int rows, int n)
{
    int r = blockIdx.x * blockDim.x + threadIdx.x;
    if (r >= rows) return;
    float mx = in[(size_t)r * n];
    for (int j = 1; j < n; ++j) mx = fmaxf(mx, in[(size_t)r * n + j]);
    float e[8], s = 0.f;
    for (int j = 0; j < n; ++j) { e[j] = __expf(in[(size_t)r * n + j] - mx); s += e[j]; }
    for (int j = 0; j < n; ++j) out[(size_t)r * n + j] = e[j] / s;
}

__global__ __launch_bounds__(256) void combine_out(
    const float* __restrict__ basep, const float* __restrict__ syn,
    const float* __restrict__ red, const float* __restrict__ iw,
    float* __restrict__ out, int ND)
{
    size_t idx = (size_t)blockIdx.x * blockDim.x + threadIdx.x;
    size_t total = (size_t)128 * ND;
    if (idx >= total) return;
    int b = (int)(idx / ND);
    float w1 = iw[b * 2], w2 = iw[b * 2 + 1];
    float w0 = 1.f - w1 - w2;
    out[idx] = w0 * basep[idx] + w1 * syn[idx] + w2 * red[idx];
}

__global__ __launch_bounds__(128) void write_weights(
    const float* __restrict__ mw, const float* __restrict__ iw, float* __restrict__ ow)
{
    int b = blockIdx.x * blockDim.x + threadIdx.x;
    if (b >= 128) return;
    for (int j = 0; j < 4; ++j) ow[b * 6 + j] = mw[b * 4 + j];
    ow[b * 6 + 4] = iw[b * 2];
    ow[b * 6 + 5] = iw[b * 2 + 1];
}

// ---------------------------------------------------------------------------
// Host-side drivers
// ---------------------------------------------------------------------------
static inline int cdiv(long long a, long long b) { return (int)((a + b - 1) / b); }

template <int EPI>
static void gemm(const float* A, int lda, const float* W, const float* bias,
                 float* C, int ldc, int M, int N, int K, hipStream_t s)
{
    dim3 g(cdiv(N, BN), cdiv(M, BM));
    gemm_wmma<EPI><<<g, dim3(128), 0, s>>>(A, lda, W, bias, C, ldc, M, N, K);
}

struct MambaP {
    const float *in_proj, *conv_w, *conv_b, *x_proj, *dt_w, *dt_b, *A_log, *D, *out_proj;
};
struct ExpertP {
    MambaP m1, m2;
    const float *c1w, *c1b, *c2w, *c2b, *ln1w, *ln1b, *ln2w, *ln2b;
};
struct WS {
    float *xs, *z, *xc, *dtv, *ym, *dbl;
    float *flipb, *m1o, *m2t, *m2o, *x1, *ffn, *y2, *actb, *eout;
    float *base, *syn, *redin, *redo, *feats;
    float *qkv, *atto, *att2, *imph, *impg, *msc, *mw, *g1, *gsc, *iw;
};

static void run_mamba(const MambaP& p, const float* act, float* out,
                      int Bc, int L, int d_in, int d_conv, WS& ws, hipStream_t s)
{
    const int T = Bc * L;
    const int D = 1024;
    // in_proj split: xs = act*Wxs, z = act*Wz
    gemm<EPI_NONE>(act, D, p.in_proj, nullptr, ws.xs, d_in, T, d_in, D, s);
    gemm<EPI_NONE>(act, D, p.in_proj + (size_t)d_in * D, nullptr, ws.z, d_in, T, d_in, D, s);
    // causal depthwise conv + silu
    {
        size_t tot = (size_t)T * d_in;
        dwconv_silu<<<cdiv(tot, 256), 256, 0, s>>>(ws.xs, p.conv_w, p.conv_b, ws.xc, Bc, L, d_in, d_conv);
    }
    // x_proj -> [T,96] (dt rank 64 | B 16 | C 16)
    gemm<EPI_NONE>(ws.xc, d_in, p.x_proj, nullptr, ws.dbl, 96, T, 96, d_in, s);
    // dt = softplus(dbl[:, :64] @ dt_w^T + dt_b)
    gemm<EPI_SOFTPLUS>(ws.dbl, 96, p.dt_w, p.dt_b, ws.dtv, d_in, T, d_in, 64, s);
    // selective scan + D skip + silu(z) gate
    mamba_scan<<<cdiv((size_t)Bc * d_in, 256), 256, 0, s>>>(
        ws.dtv, ws.dbl, ws.xc, ws.z, p.A_log, p.D, ws.ym, Bc, L, d_in);
    // out_proj
    gemm<EPI_NONE>(ws.ym, d_in, p.out_proj, nullptr, out, D, T, D, d_in, s);
}

static void run_expert(const ExpertP& p, const float* act, float* outp,
                       int Bc, int L, int d_in, int d_conv, int d_ff, WS& ws, hipStream_t s)
{
    const int T = Bc * L;
    const int D = 1024;
    const size_t TD = (size_t)T * D;
    run_mamba(p.m1, act, ws.m1o, Bc, L, d_in, d_conv, ws, s);
    flip_seq<<<cdiv(TD, 256), 256, 0, s>>>(ws.flipb, act, Bc, L, D);
    run_mamba(p.m2, ws.flipb, ws.m2t, Bc, L, d_in, d_conv, ws, s);
    flip_seq<<<cdiv(TD, 256), 256, 0, s>>>(ws.m2o, ws.m2t, Bc, L, D);
    // x1 = LN(act + m1o + m2o)
    add_ln<0><<<T, 256, 0, s>>>(act, ws.m1o, ws.m2o, p.ln1w, p.ln1b, ws.x1, D);
    // FFN: gelu(x1 @ c1w^T + c1b) @ c2w^T + c2b
    gemm<EPI_GELU>(ws.x1, D, p.c1w, p.c1b, ws.ffn, d_ff, T, d_ff, D, s);
    gemm<EPI_NONE>(ws.ffn, d_ff, p.c2w, p.c2b, ws.y2, D, T, D, d_ff, s);
    add_ln<0><<<T, 256, 0, s>>>(ws.x1, ws.y2, nullptr, p.ln2w, p.ln2b, outp, D);
}

// ---------------------------------------------------------------------------
extern "C" void kernel_launch(void* const* d_in, const int* in_sizes, int n_in,
                              void* d_out, int out_size, void* d_ws, size_t ws_size,
                              hipStream_t stream)
{
    (void)in_sizes; (void)n_in; (void)out_size; (void)ws_size;
    const int D = 1024;

    // ---- parse inputs (setup_inputs dict order, recursive insertion order)
    int ii = 0;
    auto F = [&](void) { return (const float*)d_in[ii++]; };
    const float* X = F(); // [128, 39, 1024]

    auto getM = [&](MambaP& m) {
        m.in_proj = F(); m.conv_w = F(); m.conv_b = F(); m.x_proj = F();
        m.dt_w = F(); m.dt_b = F(); m.A_log = F(); m.D = F(); m.out_proj = F();
    };
    auto getE = [&](ExpertP& e) {
        getM(e.m1); getM(e.m2);
        e.c1w = F(); e.c1b = F(); e.c2w = F(); e.c2b = F();
        e.ln1w = F(); e.ln1b = F(); e.ln2w = F(); e.ln2b = F();
    };
    ExpertP fire, weather, terrain, modis, synergy, redund;
    getE(fire); getE(weather); getE(terrain); getE(modis); getE(synergy); getE(redund);
    const float *mha_inw = F(), *mha_inb = F(), *mha_ow = F(), *mha_ob = F();
    const float *imp_w1 = F(), *imp_b1 = F(), *imp_lnw = F(), *imp_lnb = F(), *imp_w2 = F(), *imp_b2 = F();
    const float *g_w1 = F(), *g_b1 = F(), *g_w2 = F(), *g_b2 = F();

    // ---- workspace bump allocator
    const size_t TMAX = 128 * 39, DINMAX = 3072, DFFMAX = 6144;
    float* wsp = (float*)d_ws;
    size_t off = 0;
    auto alloc = [&](size_t n) { float* p = wsp + off; off += (n + 3) & ~(size_t)3; return p; };
    WS ws;
    ws.xs  = alloc(TMAX * DINMAX); ws.z   = alloc(TMAX * DINMAX);
    ws.xc  = alloc(TMAX * DINMAX); ws.dtv = alloc(TMAX * DINMAX);
    ws.ym  = alloc(TMAX * DINMAX); ws.dbl = alloc(TMAX * 96);
    ws.flipb = alloc(TMAX * D); ws.m1o = alloc(TMAX * D);
    ws.m2t = alloc(TMAX * D);   ws.m2o = alloc(TMAX * D);
    ws.x1  = alloc(TMAX * D);   ws.ffn = alloc(TMAX * DFFMAX);
    ws.y2  = alloc(TMAX * D);   ws.actb = alloc(TMAX * D);
    ws.eout = alloc(TMAX * D);
    ws.base = alloc(TMAX * D);  ws.syn = alloc(TMAX * D);
    ws.redin = alloc(TMAX * D); ws.redo = alloc(TMAX * D);
    ws.feats = alloc((size_t)128 * 4 * D);
    ws.qkv = alloc((size_t)512 * 3 * D);
    ws.atto = alloc((size_t)512 * D); ws.att2 = alloc((size_t)512 * D);
    ws.imph = alloc((size_t)512 * 512); ws.impg = alloc((size_t)512 * 512);
    ws.msc = alloc(512); ws.mw = alloc(512);
    ws.g1 = alloc((size_t)128 * D); ws.gsc = alloc(256); ws.iw = alloc(256);

    // ---- leaf experts -> base[B,39,D]
    struct { const ExpertP* p; int s0, L, d_in, d_conv, d_ff; } leaf[4] = {
        { &fire,    0,  1, 1024, 2, 1024 },
        { &weather, 1, 12, 2048, 4, 4096 },
        { &terrain, 13, 7, 2048, 4, 2048 },
        { &modis,  20, 19, 2048, 4, 3072 },
    };
    for (int e = 0; e < 4; ++e) {
        int L = leaf[e].L, s0 = leaf[e].s0;
        size_t tot = (size_t)128 * L * D;
        gather_slice<<<cdiv(tot, 256), 256, 0, stream>>>(ws.actb, X, s0, L, D);
        run_expert(*leaf[e].p, ws.actb, ws.eout, 128, L, leaf[e].d_in, leaf[e].d_conv, leaf[e].d_ff, ws, stream);
        scatter_slice<<<cdiv(tot, 256), 256, 0, stream>>>(ws.base, ws.eout, s0, L, D);
    }

    // ---- synergy expert on full base
    run_expert(synergy, ws.base, ws.syn, 128, 39, 3072, 4, 6144, ws, stream);

    // ---- feats = per-modality means of base slices
    for (int e = 0; e < 4; ++e)
        mean_slice<<<cdiv((size_t)128 * D, 256), 256, 0, stream>>>(
            ws.base, ws.feats, e, leaf[e].s0, leaf[e].L, D);

    // ---- redundancy expert
    red_prep<<<128, 256, 0, stream>>>(ws.feats, ws.redin, D);
    run_expert(redund, ws.redin, ws.redo, 128, 39, 3072, 4, 6144, ws, stream);

    // ---- MHA over feats [B,4,D]
    gemm<EPI_NONE>(ws.feats, D, mha_inw, mha_inb, ws.qkv, 3 * D, 512, 3 * D, D, stream);
    mha_attn<<<128 * 8, 128, 0, stream>>>(ws.qkv, ws.atto, D, 8, 128);
    gemm<EPI_NONE>(ws.atto, D, mha_ow, mha_ob, ws.att2, D, 512, D, D, stream);

    // ---- importance head: gelu(LN(att2 @ w1^T + b1)) @ w2^T + b2, softmax
    gemm<EPI_NONE>(ws.att2, D, imp_w1, imp_b1, ws.imph, 512, 512, 512, D, stream);
    add_ln<1><<<512, 256, 0, stream>>>(ws.imph, nullptr, nullptr, imp_lnw, imp_lnb, ws.impg, 512);
    small_linear<<<cdiv(512, 256), 256, 0, stream>>>(ws.impg, 512, imp_w2, imp_b2, ws.msc, 512, 1, 512);
    row_softmax<<<1, 128, 0, stream>>>(ws.msc, ws.mw, 128, 4);

    // ---- gate: relu(flat @ w1^T + b1) @ w2^T + b2, softmax
    gemm<EPI_RELU>(ws.feats, 4 * D, g_w1, g_b1, ws.g1, D, 128, D, 4 * D, stream);
    small_linear<<<1, 256, 0, stream>>>(ws.g1, D, g_w2, g_b2, ws.gsc, 128, 2, D);
    row_softmax<<<1, 128, 0, stream>>>(ws.gsc, ws.iw, 128, 2);

    // ---- final combine + weights
    float* out = (float*)d_out;
    const int ND = 39 * D;
    combine_out<<<cdiv((size_t)128 * ND, 256), 256, 0, stream>>>(
        ws.base, ws.syn, ws.redo, ws.iw, out, ND);
    write_weights<<<1, 128, 0, stream>>>(ws.mw, ws.iw, out + (size_t)128 * ND);
}
